// NTMCell_31396210933878
// MI455X (gfx1250) — matrix operand order repaired and checked by reading, not compiled, not run on hardware
//
#include <hip/hip_runtime.h>
#include <math.h>

// ---------------------------------------------------------------------------
// NTM cell for MI455X (gfx1250, wave32). bf16 WMMA for the three GEMMs with
// N-major (transposed) weights so every fragment is a contiguous 256-bit
// vector load; single-pass LDS-staged addressing/memory-update kernel.
// ---------------------------------------------------------------------------

typedef __bf16 bf16;
typedef __attribute__((ext_vector_type(16))) __bf16 v16bf;
typedef __attribute__((ext_vector_type(8)))  __bf16 v8bf;
typedef __attribute__((ext_vector_type(8)))  float  v8f;

#define B_N     4096
#define U_N     256
#define MEM_N   128
#define W_N     64
#define OUT_N   64
#define NPH     70
#define KZ      384     // IN + W + U  (concat [x|rv0|h0])
#define NZ      1024    // 4*U
#define NP_PAD  272     // params cols padded to mult of 16
#define NP_REAL 268
#define KO      320     // U + W (concat [h_new|read_vec])
#define CLIPV   20.0f
#define EPSV    1e-12f

// workspace layout (bytes), all 256B aligned; total ~11.2 MB
#define WZT_OFF 0u          // bf16 [1024,384]  (Wz transposed, N-major)
#define WPT_OFF 786432u     // bf16 [272,256]   (Wp transposed, zero-padded rows 268..271)
#define WOT_OFF 925696u     // bf16 [64,320]    (Wo transposed)
#define CI_OFF  966656u     // bf16 [4096,384]
#define AO_OFF  4112384u    // bf16 [4096,320]
#define PAR_OFF 6733824u    // f32  [4096,272]

__device__ __forceinline__ float sigmoidf_(float x){ return 1.0f/(1.0f+expf(-x)); }
__device__ __forceinline__ float softplusf_(float x){ return (x > 20.f) ? x : log1pf(expf(x)); }
__device__ __forceinline__ float clipf_(float x){ return fminf(fmaxf(x, -CLIPV), CLIPV); }

__device__ __forceinline__ v8f wmma_bf16(v16bf a, v16bf b, v8f c){
  // v_wmma_f32_16x16x32_bf16: (neg_a, A, neg_b, B, c_mod, C, reuse_a, reuse_b)
  return __builtin_amdgcn_wmma_f32_16x16x32_bf16(false, a, false, b, (short)0, c, false, false);
}

// A fragment, 16x32 bf16 (MxK), row-major source with leading dim lda.
// Layout used here: lane<16 holds M=lane, K=0..15; lane>=16 holds K=16..31 (matches B).
// Two contiguous 16-byte vector loads + shuffle (no element movs).
__device__ __forceinline__ v16bf load_a_frag(const bf16* A, int lda, int k0, int lane){
  const int m = lane & 15, h = lane >> 4;
  const bf16* p = A + (size_t)m*lda + k0 + h*16;
  v8bf lo = *(const v8bf*)p;
  v8bf hi = *(const v8bf*)(p + 8);
  return __builtin_shufflevector(lo, hi, 0,1,2,3,4,5,6,7,8,9,10,11,12,13,14,15);
}

// B fragment, logical 32x16 (KxN) but stored N-major (Bt[n][k], leading dim ldk).
// Layout used here: N = lane&15; half-wave 0 holds K=0..15, half 1 K=16..31 ->
// per lane this is 16 CONTIGUOUS bf16 = one 32-byte vector load.
__device__ __forceinline__ v16bf load_b_frag(const bf16* Bt, int ldk, int k0, int n0, int lane){
  const int n = lane & 15, h = lane >> 4;
  return *(const v16bf*)(Bt + (size_t)(n0 + n)*ldk + k0 + 16*h);
}

// ---------------------------------------------------------------------------
// Kernel 0: pack activations + transpose/convert weights to bf16
// ---------------------------------------------------------------------------
__global__ void convert_kernel(const float* __restrict__ x, const float* __restrict__ rv0,
                               const float* __restrict__ h0,
                               const float* __restrict__ W_k, const float* __restrict__ W_r,
                               const float* __restrict__ W_p, const float* __restrict__ W_o,
                               bf16* __restrict__ CI, bf16* __restrict__ WzT,
                               bf16* __restrict__ WpT, bf16* __restrict__ WoT)
{
  const int NCI = B_N*KZ, NWZ = NZ*KZ, NWP = NP_PAD*U_N, NWO = OUT_N*KO;
  int i = blockIdx.x*blockDim.x + threadIdx.x;
  if (i >= NCI + NWZ + NWP + NWO) return;
  if (i < NCI){
    int bb = i / KZ, k = i % KZ;
    float v = (k < 64) ? x[bb*64 + k] : (k < 128) ? rv0[bb*64 + (k-64)] : h0[bb*256 + (k-128)];
    CI[i] = (bf16)v;
  } else if (i < NCI + NWZ){
    int j = i - NCI; int n = j / KZ, k = j % KZ;     // WzT[n][k]
    float v = (k < 128) ? W_k[k*NZ + n] : W_r[(k-128)*NZ + n];
    WzT[j] = (bf16)v;
  } else if (i < NCI + NWZ + NWP){
    int j = i - NCI - NWZ; int n = j / U_N, k = j % U_N;  // WpT[n][k]
    WpT[j] = (bf16)((n < NP_REAL) ? W_p[k*NP_REAL + n] : 0.f);
  } else {
    int j = i - NCI - NWZ - NWP; int n = j / KO, k = j % KO;  // WoT[n][k]
    WoT[j] = (bf16)W_o[k*OUT_N + n];
  }
}

// ---------------------------------------------------------------------------
// Kernel 1: z = CI @ Wz (+b), fused LSTM nonlinearity in registers.
// grid = B/16 blocks x 256 threads (8 waves). Wave w owns u-tiles {w, w+8};
// the 4 gate column-tiles (u, u+256, u+512, u+768) all satisfy nt % 8 == w,
// so each wave holds all four gate accumulators of its columns in VGPRs.
// ---------------------------------------------------------------------------
__global__ __launch_bounds__(256)
void lstm_gemm_kernel(const bf16* __restrict__ CI, const bf16* __restrict__ WzT,
                      const float* __restrict__ b_lstm, const float* __restrict__ c0,
                      float* __restrict__ h_out, float* __restrict__ c_out,
                      bf16* __restrict__ AO)
{
  const int tid = threadIdx.x, lane = tid & 31, wave = tid >> 5;
  const int half = lane >> 4, nl = lane & 15;
  const int row0 = blockIdx.x * 16;
  const bf16* Arow = CI + (size_t)row0 * KZ;

  v16bf af[12];
  #pragma unroll
  for (int kt = 0; kt < 12; ++kt) af[kt] = load_a_frag(Arow, KZ, kt*32, lane);

  #pragma unroll
  for (int uu = 0; uu < 2; ++uu){
    const int ut = wave + uu*8;             // u-tile 0..15
    v8f ai = {}, afr = {}, ag = {}, ao = {};
    #pragma unroll
    for (int kt = 0; kt < 12; ++kt){
      v16bf b0 = load_b_frag(WzT, KZ, kt*32, ut*16,       lane);
      v16bf b1 = load_b_frag(WzT, KZ, kt*32, 256 + ut*16, lane);
      v16bf b2 = load_b_frag(WzT, KZ, kt*32, 512 + ut*16, lane);
      v16bf b3 = load_b_frag(WzT, KZ, kt*32, 768 + ut*16, lane);
      ai  = wmma_bf16(af[kt], b0, ai);
      afr = wmma_bf16(af[kt], b1, afr);
      ag  = wmma_bf16(af[kt], b2, ag);
      ao  = wmma_bf16(af[kt], b3, ao);
    }
    const int u = ut*16 + nl;
    const float bi = b_lstm[u], bfv = b_lstm[256+u], bg = b_lstm[512+u], bo = b_lstm[768+u];
    #pragma unroll
    for (int r = 0; r < 8; ++r){
      const int gb = row0 + r + 8*half;     // C/D layout: M = r + 8*half
      float zi = ai[r]+bi, zf = afr[r]+bfv, zg = ag[r]+bg, zo = ao[r]+bo;
      float cn = sigmoidf_(zf)*c0[(size_t)gb*U_N + u] + sigmoidf_(zi)*tanhf(zg);
      float hn = sigmoidf_(zo)*tanhf(cn);
      c_out[(size_t)gb*U_N + u] = cn;
      h_out[(size_t)gb*U_N + u] = hn;
      AO[(size_t)gb*KO + u] = (bf16)hn;     // concat buffer cols 0..255
    }
  }
}

// ---------------------------------------------------------------------------
// Kernel 2: params = clip(h_new @ W_p + b_p). K=256, N=272(pad of 268).
// ---------------------------------------------------------------------------
__global__ __launch_bounds__(128)
void params_gemm_kernel(const bf16* __restrict__ AO, const bf16* __restrict__ WpT,
                        const float* __restrict__ b_p, float* __restrict__ params)
{
  const int tid = threadIdx.x, lane = tid & 31, wave = tid >> 5;
  const int half = lane >> 4, nl = lane & 15;
  const int row0 = blockIdx.x * 16;
  const bf16* Arow = AO + (size_t)row0 * KO;

  v16bf af[8];
  #pragma unroll
  for (int kt = 0; kt < 8; ++kt) af[kt] = load_a_frag(Arow, KO, kt*32, lane);

  for (int nt = wave; nt < 17; nt += 4){
    v8f acc = {};
    #pragma unroll
    for (int kt = 0; kt < 8; ++kt){
      v16bf bfrag = load_b_frag(WpT, U_N, kt*32, nt*16, lane);
      acc = wmma_bf16(af[kt], bfrag, acc);
    }
    const int c = nt*16 + nl;
    if (c < NP_REAL){
      const float bias = b_p[c];
      #pragma unroll
      for (int r = 0; r < 8; ++r)
        params[(size_t)(row0 + r + 8*half)*NP_PAD + c] = clipf_(acc[r] + bias);
    }
  }
}

// ---------------------------------------------------------------------------
// Kernel 3: addressing + read + memory update. One workgroup per batch row;
// M[b] staged once in LDS, so M is read from HBM exactly once and M_new
// written once (this is the HBM roofline term of the whole cell).
// ---------------------------------------------------------------------------
__device__ __forceinline__ float blockReduceSum128(float v, float* red, int tid){
  red[tid] = v; __syncthreads();
  #pragma unroll
  for (int s = 64; s > 0; s >>= 1){ if (tid < s) red[tid] += red[tid + s]; __syncthreads(); }
  float r = red[0]; __syncthreads(); return r;
}
__device__ __forceinline__ float blockReduceMax128(float v, float* red, int tid){
  red[tid] = v; __syncthreads();
  #pragma unroll
  for (int s = 64; s > 0; s >>= 1){ if (tid < s) red[tid] = fmaxf(red[tid], red[tid + s]); __syncthreads(); }
  float r = red[0]; __syncthreads(); return r;
}

__global__ __launch_bounds__(128)
void ntm_update_kernel(const float* __restrict__ M, const float* __restrict__ params,
                       const float* __restrict__ w0_prev, const float* __restrict__ w1_prev,
                       float* __restrict__ out_rv, float* __restrict__ out_wr,
                       float* __restrict__ out_ww, float* __restrict__ out_Mnew,
                       bf16* __restrict__ AO)
{
  __shared__ float Msh[MEM_N*68];     // 128 rows, stride 68 (16B-aligned, conflict pad)
  __shared__ float hp[2*NPH];
  __shared__ float kv[2][W_N];
  __shared__ float rn[MEM_N];
  __shared__ float wgsh[MEM_N];
  __shared__ float wfin[2][MEM_N];
  __shared__ float er[W_N], ad[W_N];
  __shared__ float sc[2][8];          // knfac,beta,g,s0,s1,s2,gamma
  __shared__ float red[MEM_N];

  const int tid = threadIdx.x;
  const int b = blockIdx.x;
  const float*  Mb = M + (size_t)b*MEM_N*W_N;
  const float*  pb = params + (size_t)b*NP_PAD;

  // stage M[b] (float4)
  const float4* M4 = (const float4*)Mb;
  for (int i = tid; i < MEM_N*W_N/4; i += 128){
    const int n = i >> 4, w4 = i & 15;
    float4 v = M4[i];
    float* dst = &Msh[n*68 + w4*4];
    dst[0]=v.x; dst[1]=v.y; dst[2]=v.z; dst[3]=v.w;
  }
  for (int i = tid; i < 2*NPH; i += 128) hp[i] = pb[i];
  if (tid < W_N){
    er[tid] = sigmoidf_(pb[2*NPH + tid]);
    ad[tid] = tanhf(pb[2*NPH + W_N + tid]);
  }
  __syncthreads();

  // per-slot 1/||M[b,n,:]||
  {
    float s = 0.f;
    #pragma unroll 8
    for (int w = 0; w < W_N; ++w){ float m = Msh[tid*68 + w]; s += m*m; }
    rn[tid] = rsqrtf(fmaxf(s, EPSV));
  }

  for (int hsel = 0; hsel < 2; ++hsel){
    const float* hpp = hp + hsel*NPH;
    if (tid < W_N) kv[hsel][tid] = tanhf(hpp[tid]);
    __syncthreads();
    if (tid == 0){
      float s = 0.f;
      for (int w = 0; w < W_N; ++w){ float k = kv[hsel][w]; s += k*k; }
      sc[hsel][0] = rsqrtf(fmaxf(s, EPSV));       // key norm factor
      sc[hsel][1] = softplusf_(hpp[64]);          // beta
      sc[hsel][2] = sigmoidf_(hpp[65]);           // g
      float e0 = hpp[66], e1 = hpp[67], e2 = hpp[68];
      float mx = fmaxf(e0, fmaxf(e1, e2));
      float x0 = expf(e0-mx), x1 = expf(e1-mx), x2 = expf(e2-mx);
      float si = 1.f/(x0+x1+x2);
      sc[hsel][3] = x0*si; sc[hsel][4] = x1*si; sc[hsel][5] = x2*si;
      sc[hsel][6] = softplusf_(hpp[69]) + 1.f;    // gamma
    }
    __syncthreads();
    const float knf = sc[hsel][0], beta = sc[hsel][1], g = sc[hsel][2];
    const float s0 = sc[hsel][3], s1 = sc[hsel][4], s2 = sc[hsel][5], gamma = sc[hsel][6];

    // content similarity (ref keeps NEGATIVE cosine similarity)
    float dot = 0.f;
    #pragma unroll 8
    for (int w = 0; w < W_N; ++w) dot += kv[hsel][w] * Msh[tid*68 + w];
    float a = beta * (-(dot * knf * rn[tid]));
    float amax = blockReduceMax128(a, red, tid);
    float e = expf(a - amax);
    float esum = blockReduceSum128(e, red, tid);
    float wc = e / esum;

    const float* prevw = (hsel == 0) ? w0_prev : w1_prev;
    float wgv = g*wc + (1.f - g)*prevw[(size_t)b*MEM_N + tid];
    wgsh[tid] = wgv;
    __syncthreads();
    // 3-tap circular shift:  w_[i] = s0*wg[i] + s1*wg[i-1] + s2*wg[i+1]
    float wsf = s0*wgsh[tid] + s1*wgsh[(tid+127)&127] + s2*wgsh[(tid+1)&127];
    float wpw = powf(wsf, gamma);
    float wsum = blockReduceSum128(wpw, red, tid);
    float wf = wpw / wsum;
    wfin[hsel][tid] = wf;
    if (hsel == 0) out_wr[(size_t)b*MEM_N + tid] = wf;
    else           out_ww[(size_t)b*MEM_N + tid] = wf;
    __syncthreads();
  }

  // read vector: rv[w] = sum_n w_read[n] * M[b,n,w]
  if (tid < W_N){
    float r = 0.f;
    #pragma unroll 8
    for (int n = 0; n < MEM_N; ++n) r += wfin[0][n] * Msh[n*68 + tid];
    out_rv[(size_t)b*W_N + tid] = r;
    AO[(size_t)b*KO + U_N + tid] = (bf16)r;       // concat buffer cols 256..319
  }

  // memory update: M_new = M*(1 - ww*erase) + ww*add
  {
    const float ww = wfin[1][tid];
    float4* out4 = (float4*)(out_Mnew + (size_t)b*MEM_N*W_N + (size_t)tid*W_N);
    #pragma unroll
    for (int w4 = 0; w4 < 16; ++w4){
      const float* m = &Msh[tid*68 + w4*4];
      float4 o;
      o.x = m[0]*(1.f - ww*er[w4*4+0]) + ww*ad[w4*4+0];
      o.y = m[1]*(1.f - ww*er[w4*4+1]) + ww*ad[w4*4+1];
      o.z = m[2]*(1.f - ww*er[w4*4+2]) + ww*ad[w4*4+2];
      o.w = m[3]*(1.f - ww*er[w4*4+3]) + ww*ad[w4*4+3];
      out4[w4] = o;
    }
  }
}

// ---------------------------------------------------------------------------
// Kernel 4: out = clip([h_new|read_vec] @ W_o + b_o). K=320, N=64.
// ---------------------------------------------------------------------------
__global__ __launch_bounds__(128)
void out_gemm_kernel(const bf16* __restrict__ AO, const bf16* __restrict__ WoT,
                     const float* __restrict__ b_o, float* __restrict__ out)
{
  const int tid = threadIdx.x, lane = tid & 31, wave = tid >> 5;
  const int half = lane >> 4, nl = lane & 15;
  const int row0 = blockIdx.x * 16;
  const bf16* Arow = AO + (size_t)row0 * KO;

  v16bf af[10];
  #pragma unroll
  for (int kt = 0; kt < 10; ++kt) af[kt] = load_a_frag(Arow, KO, kt*32, lane);

  const int nt = wave;                 // 4 waves cover N=64
  v8f acc = {};
  #pragma unroll
  for (int kt = 0; kt < 10; ++kt){
    v16bf bfrag = load_b_frag(WoT, KO, kt*32, nt*16, lane);
    acc = wmma_bf16(af[kt], bfrag, acc);
  }
  const int c = nt*16 + nl;
  const float bias = b_o[c];
  #pragma unroll
  for (int r = 0; r < 8; ++r)
    out[(size_t)(row0 + r + 8*half)*OUT_N + c] = clipf_(acc[r] + bias);
}

// ---------------------------------------------------------------------------
extern "C" void kernel_launch(void* const* d_in, const int* in_sizes, int n_in,
                              void* d_out, int out_size, void* d_ws, size_t ws_size,
                              hipStream_t stream)
{
  (void)in_sizes; (void)n_in; (void)out_size; (void)ws_size;
  const float* x   = (const float*)d_in[0];
  const float* h0  = (const float*)d_in[1];
  const float* c0  = (const float*)d_in[2];
  const float* rv0 = (const float*)d_in[3];
  const float* w0p = (const float*)d_in[4];
  const float* w1p = (const float*)d_in[5];
  const float* M   = (const float*)d_in[6];
  const float* W_k = (const float*)d_in[7];
  const float* W_r = (const float*)d_in[8];
  const float* b_l = (const float*)d_in[9];
  const float* W_p = (const float*)d_in[10];
  const float* b_p = (const float*)d_in[11];
  const float* W_o = (const float*)d_in[12];
  const float* b_o = (const float*)d_in[13];

  char* ws = (char*)d_ws;
  bf16*  WzT    = (bf16*)(ws + WZT_OFF);
  bf16*  WpT    = (bf16*)(ws + WPT_OFF);
  bf16*  WoT    = (bf16*)(ws + WOT_OFF);
  bf16*  CI     = (bf16*)(ws + CI_OFF);
  bf16*  AO     = (bf16*)(ws + AO_OFF);
  float* params = (float*)(ws + PAR_OFF);

  float* out    = (float*)d_out;               // tuple, concatenated flat:
  float* o_out  = out;                         // out      [4096,64]
  float* o_rv   = out + 262144;                // read_vec [4096,64]
  float* o_wr   = out + 524288;                // w_read   [4096,128]
  float* o_ww   = out + 1048576;               // w_write  [4096,128]
  float* o_Mn   = out + 1572864;               // M_new    [4096,128,64]
  float* o_h    = out + 35127296;              // h_new    [4096,256]
  float* o_c    = out + 36175872;              // c_new    [4096,256]

  const int total = B_N*KZ + NZ*KZ + NP_PAD*U_N + OUT_N*KO;
  convert_kernel    <<<(total + 255)/256, 256, 0, stream>>>(x, rv0, h0, W_k, W_r, W_p, W_o,
                                                            CI, WzT, WpT, WoT);
  lstm_gemm_kernel  <<<B_N/16, 256, 0, stream>>>(CI, WzT, b_l, c0, o_h, o_c, AO);
  params_gemm_kernel<<<B_N/16, 128, 0, stream>>>(AO, WpT, b_p, params);
  ntm_update_kernel <<<B_N,    128, 0, stream>>>(M, params, w0p, w1p, o_rv, o_wr, o_ww, o_Mn, AO);
  out_gemm_kernel   <<<B_N/16, 128, 0, stream>>>(AO, WoT, b_o, o_out);
}
